// MultiHeadSelfAttention_21492016349331
// MI455X (gfx1250) — matrix operand order repaired and checked by reading, not compiled, not run on hardware
//
#include <hip/hip_runtime.h>
#include <hip/hip_bf16.h>
#include <math.h>
#include <stdint.h>

typedef __attribute__((ext_vector_type(16))) __bf16 v16bf;
typedef __attribute__((ext_vector_type(8)))  float  v8f;

#define NB     4
#define SEQ    2048
#define DMODEL 1024
#define NH     16
#define DK     64
// log2(10000) / (DK/2)
#define NEG_L2T_OVER_HALFDK (-13.2877123795494f / 32.0f)

// ---------------- async global->LDS copy (ASYNCcnt path), with fallback ----------------
#if defined(__AMDGCN__) && __has_builtin(__builtin_amdgcn_global_load_async_to_lds_b128)
#define HAS_ASYNC 1
#else
#define HAS_ASYNC 0
#endif

typedef int v4i_vec __attribute__((vector_size(16)));
#if HAS_ASYNC
typedef __attribute__((address_space(1))) v4i_vec* g_v4i_p;
typedef __attribute__((address_space(3))) v4i_vec* l_v4i_p;
#endif

__device__ __forceinline__ void async_copy16(const __bf16* gsrc, __bf16* ldst) {
#if HAS_ASYNC
  __builtin_amdgcn_global_load_async_to_lds_b128(
      (g_v4i_p)(uintptr_t)gsrc, (l_v4i_p)(uintptr_t)ldst, 0, 0);
#else
  *(uint4*)ldst = *(const uint4*)gsrc;
#endif
}

__device__ __forceinline__ void async_wait0() {
#if HAS_ASYNC
#if __has_builtin(__builtin_amdgcn_s_wait_asynccnt)
  __builtin_amdgcn_s_wait_asynccnt(0);
#else
  asm volatile("s_wait_asynccnt 0x0" ::: "memory");
#endif
#endif
}

// ---------------- WMMA fragment loaders (contiguous k order: k = half*16 + e) ----------------
// A 16x32 bf16: lane<16 -> row M=lane, elem e -> K = k0+half*16+e
__device__ __forceinline__ v16bf frag_A(const __bf16* row0, int stride, int k0, int lane) {
  const int half = (lane >> 4) & 1;
  const __bf16* p = row0 + (lane & 15) * stride + k0 + half * 16;
  v16bf a;
#pragma unroll
  for (int e = 0; e < 16; ++e) a[e] = p[e];
  return a;
}
// B 32x16 bf16, B stored transposed in LDS as [n][k]: elem e -> K = k0+half*16+e, N = lane&15
__device__ __forceinline__ v16bf frag_BT(const __bf16* base, int stride, int n0, int k0, int lane) {
  const int half = (lane >> 4) & 1;
  const __bf16* p = base + (n0 + (lane & 15)) * stride + k0 + half * 16;
  v16bf b;
#pragma unroll
  for (int e = 0; e < 16; ++e) b[e] = p[e];
  return b;
}
// B 32x16 bf16, B stored natural [k][n] in LDS
__device__ __forceinline__ v16bf frag_BK(const __bf16* base, int stride, int n0, int k0, int lane) {
  const int half = (lane >> 4) & 1;
  const int n = n0 + (lane & 15);
  v16bf b;
#pragma unroll
  for (int e = 0; e < 16; ++e) b[e] = base[(k0 + half * 16 + e) * stride + n];
  return b;
}

// ---------------- fp32 -> bf16 conversion prepass ----------------
__global__ __launch_bounds__(256) void cvt_bf16_kernel(const float* __restrict__ src,
                                                       __bf16* __restrict__ dst, int n) {
  const int i = (blockIdx.x * 256 + threadIdx.x) * 8;
  if (i < n) {
    float4 a = *(const float4*)(src + i);
    float4 b = *(const float4*)(src + i + 4);
    dst[i + 0] = (__bf16)a.x; dst[i + 1] = (__bf16)a.y;
    dst[i + 2] = (__bf16)a.z; dst[i + 3] = (__bf16)a.w;
    dst[i + 4] = (__bf16)b.x; dst[i + 5] = (__bf16)b.y;
    dst[i + 6] = (__bf16)b.z; dst[i + 7] = (__bf16)b.w;
  }
}

// ---------------- double-buffered bf16 GEMM: C[m,n] = sum_i A[m,i] * B[n,i] ----------------
#define BM 128
#define BN 128
#define BK 32
#define LDT 40   // padded LDS stride (80B rows -> 16B aligned)

// ROPE_LAYOUT: true -> bf16 head-major [b,h,s,dk] with optional RoPE; false -> fp32 row-major
template <bool ROPE_LAYOUT>
__global__ __launch_bounds__(256) void gemm_bf16_kernel(
    const __bf16* __restrict__ A, const __bf16* __restrict__ B,
    __bf16* __restrict__ OutBf, float* __restrict__ OutF, int do_rope) {
  __shared__ __bf16 As[2][BM][LDT];
  __shared__ __bf16 Bs[2][BN][LDT];
  const int tid = threadIdx.x, lane = tid & 31, wave = tid >> 5;
  const int wm = wave & 3, wn = wave >> 2;          // 4(M) x 2(N) wave grid
  const int m0 = blockIdx.x * BM, n0 = blockIdx.y * BN;
  const int half = (lane >> 4) & 1, l15 = lane & 15;

  v8f zero = {};
  v8f acc[2][4];
#pragma unroll
  for (int mi = 0; mi < 2; ++mi)
#pragma unroll
    for (int ni = 0; ni < 4; ++ni) acc[mi][ni] = zero;

  // tile loader: 128x32 bf16 per matrix = 512 x 16B chunks; 256 threads x 2 chunks
  auto load_tiles = [&](int buf, int k0) {
#pragma unroll
    for (int ph = 0; ph < 2; ++ph) {
      const int ch = tid + ph * 256;
      const int r = ch >> 2, c = (ch & 3) * 8;
      async_copy16(A + (size_t)(m0 + r) * DMODEL + k0 + c, &As[buf][r][c]);
      async_copy16(B + (size_t)(n0 + r) * DMODEL + k0 + c, &Bs[buf][r][c]);
    }
  };

  load_tiles(0, 0);
  async_wait0();
  __syncthreads();

  int buf = 0;
  for (int k0 = 0; k0 < DMODEL; k0 += BK) {
    if (k0 + BK < DMODEL) load_tiles(buf ^ 1, k0 + BK);   // prefetch next tile

    v16bf afr[2], bfr[4];
#pragma unroll
    for (int mi = 0; mi < 2; ++mi)
      afr[mi] = frag_A(&As[buf][wm * 32 + mi * 16][0], LDT, 0, lane);
#pragma unroll
    for (int ni = 0; ni < 4; ++ni)
      bfr[ni] = frag_BT(&Bs[buf][0][0], LDT, wn * 64 + ni * 16, 0, lane);
#pragma unroll
    for (int mi = 0; mi < 2; ++mi)
#pragma unroll
      for (int ni = 0; ni < 4; ++ni)
        acc[mi][ni] = __builtin_amdgcn_wmma_f32_16x16x32_bf16(
            false, afr[mi], false, bfr[ni], (short)0, acc[mi][ni], false, false);

    async_wait0();
    __syncthreads();
    buf ^= 1;
  }

#pragma unroll
  for (int mi = 0; mi < 2; ++mi)
#pragma unroll
    for (int ni = 0; ni < 4; ++ni)
#pragma unroll
      for (int r = 0; r < 8; ++r) {
        float v = acc[mi][ni][r];
        const int gm = m0 + wm * 32 + mi * 16 + r + half * 8;
        const int gn = n0 + wn * 64 + ni * 16 + l15;
        if (ROPE_LAYOUT) {
          float res = v;
          float part = __shfl_xor(v, 1, 32);   // adjacent-pair partner, uniform control flow
          if (do_rope) {
            const int pos = gm & (SEQ - 1);
            const int dd  = gn & (DK - 1);
            const int pr  = dd >> 1;
            const float freq = exp2f((float)pr * NEG_L2T_OVER_HALFDK);
            float si, co;
            __sincosf((float)pos * freq, &si, &co);
            res = (dd & 1) ? (part * si + v * co) : (v * co - part * si);
          }
          const int bb = gm >> 11, sidx = gm & (SEQ - 1);
          const int hh = gn >> 6,  dd2  = gn & (DK - 1);
          OutBf[(((size_t)(bb * NH + hh) * SEQ + sidx) << 6) + dd2] = (__bf16)res;
        } else {
          OutF[(size_t)gm * DMODEL + gn] = v;
        }
      }
}

// ---------------- Flash attention (causal, online softmax, double-buffered K/V) ----------------
#define QT  64
#define KTL 64
#define LKV 72   // padded: 144B rows, 16B aligned
#define LPS 72

__global__ __launch_bounds__(128) void flash_attn_kernel(
    const __bf16* __restrict__ Q, const __bf16* __restrict__ K,
    const __bf16* __restrict__ V, __bf16* __restrict__ O) {
  __shared__ __bf16 Kt[2][KTL][LKV];
  __shared__ __bf16 Vt[2][KTL][LKV];
  __shared__ __bf16 Ps[4][16][LPS];   // per-wave P staging
  const int tid = threadIdx.x, lane = tid & 31, wave = tid >> 5;
  const int qt = blockIdx.x, hh = blockIdx.y, bb = blockIdx.z;
  const int half = (lane >> 4) & 1, l15 = lane & 15;
  const size_t headoff = (size_t)(bb * NH + hh) * SEQ * DK;
  const int qrow0 = qt * QT + wave * 16;

  v16bf qf[2];
#pragma unroll
  for (int kk = 0; kk < 2; ++kk)
    qf[kk] = frag_A(Q + headoff + (size_t)qrow0 * DK, DK, kk * 32, lane);

  v8f zero = {};
  v8f ao[4];
  float m[8], l[8];
#pragma unroll
  for (int ni = 0; ni < 4; ++ni) ao[ni] = zero;
#pragma unroll
  for (int r = 0; r < 8; ++r) { m[r] = -1e30f; l[r] = 0.0f; }

  // K/V tile loader: 64x64 bf16 = 512 x 16B chunks per matrix; 128 threads x 4 chunks
  auto load_kv = [&](int buf, int j) {
#pragma unroll
    for (int ph = 0; ph < 4; ++ph) {
      const int ch = tid + ph * 128;
      const int r = ch >> 3, c = (ch & 7) * 8;
      async_copy16(K + headoff + (size_t)(j * KTL + r) * DK + c, &Kt[buf][r][c]);
      async_copy16(V + headoff + (size_t)(j * KTL + r) * DK + c, &Vt[buf][r][c]);
    }
  };

  load_kv(0, 0);
  async_wait0();
  __syncthreads();

  int buf = 0;
  for (int j = 0; j <= qt; ++j) {
    if (j < qt) load_kv(buf ^ 1, j + 1);   // prefetch next K/V tile

    // Hoist ALL fragment loads for this tile up front so the scheduler can
    // stagger s_wait_dscnt across the WMMA chain instead of load->wait0->wmma
    // lockstep. V fragments are consumed only after the softmax, giving the
    // strided frag_BK reads ~100+ instructions of latency cover.
    v16bf bk[4][2], bv[2][4];
#pragma unroll
    for (int ni = 0; ni < 4; ++ni)
#pragma unroll
      for (int kk = 0; kk < 2; ++kk)
        bk[ni][kk] = frag_BT(&Kt[buf][0][0], LKV, ni * 16, kk * 32, lane);
#pragma unroll
    for (int kk = 0; kk < 2; ++kk)
#pragma unroll
      for (int ni = 0; ni < 4; ++ni)
        bv[kk][ni] = frag_BK(&Vt[buf][0][0], LKV, ni * 16, kk * 32, lane);

    // S = Q @ K^T (16 x 64 per wave)
    v8f sc[4];
#pragma unroll
    for (int ni = 0; ni < 4; ++ni) {
      sc[ni] = zero;
#pragma unroll
      for (int kk = 0; kk < 2; ++kk)
        sc[ni] = __builtin_amdgcn_wmma_f32_16x16x32_bf16(
            false, qf[kk], false, bk[ni][kk], (short)0, sc[ni], false, false);
    }

    // scale + causal mask
    float sv[4][8];
#pragma unroll
    for (int ni = 0; ni < 4; ++ni)
#pragma unroll
      for (int r = 0; r < 8; ++r) {
        const int qg = qrow0 + r + half * 8;
        const int kg = j * KTL + ni * 16 + l15;
        sv[ni][r] = (kg <= qg) ? sc[ni][r] * 0.125f : -1e30f;
      }

    // online softmax: a row spans the 16 lanes of one half -> butterfly over 1,2,4,8
#pragma unroll
    for (int r = 0; r < 8; ++r) {
      float rm = fmaxf(fmaxf(sv[0][r], sv[1][r]), fmaxf(sv[2][r], sv[3][r]));
#pragma unroll
      for (int off = 1; off < 16; off <<= 1) rm = fmaxf(rm, __shfl_xor(rm, off, 32));
      const float mnew = fmaxf(m[r], rm);
      const float corr = __expf(m[r] - mnew);
      m[r] = mnew;
      float rs = 0.0f;
#pragma unroll
      for (int ni = 0; ni < 4; ++ni) {
        const float p = __expf(sv[ni][r] - mnew);
        rs += p;
        Ps[wave][r + half * 8][ni * 16 + l15] = (__bf16)p;
      }
#pragma unroll
      for (int off = 1; off < 16; off <<= 1) rs += __shfl_xor(rs, off, 32);
      l[r] = l[r] * corr + rs;
#pragma unroll
      for (int ni = 0; ni < 4; ++ni) ao[ni][r] *= corr;
    }

    // O += P @ V (wave-private LDS restage; in-wave LDS ordering suffices)
#pragma unroll
    for (int kk = 0; kk < 2; ++kk) {
      v16bf pa = frag_A(&Ps[wave][0][0], LPS, kk * 32, lane);
#pragma unroll
      for (int ni = 0; ni < 4; ++ni)
        ao[ni] = __builtin_amdgcn_wmma_f32_16x16x32_bf16(
            false, pa, false, bv[kk][ni], (short)0, ao[ni], false, false);
    }

    async_wait0();
    __syncthreads();
    buf ^= 1;
  }

  // write [b][s][h*dk] bf16
#pragma unroll
  for (int ni = 0; ni < 4; ++ni)
#pragma unroll
    for (int r = 0; r < 8; ++r) {
      const int sidx = qrow0 + r + half * 8;
      const int col = hh * DK + ni * 16 + l15;
      O[((size_t)bb * SEQ + sidx) * DMODEL + col] = (__bf16)(ao[ni][r] / l[r]);
    }
}

// ---------------- host launch ----------------
extern "C" void kernel_launch(void* const* d_in, const int* in_sizes, int n_in,
                              void* d_out, int out_size, void* d_ws, size_t ws_size,
                              hipStream_t stream) {
  (void)in_sizes; (void)n_in; (void)out_size; (void)ws_size;
  const float* x  = (const float*)d_in[0];
  const float* Wq = (const float*)d_in[1];
  const float* Wk = (const float*)d_in[2];
  const float* Wv = (const float*)d_in[3];
  const float* Wo = (const float*)d_in[4];
  float* out = (float*)d_out;

  // workspace layout (bf16 elements):
  //   Xb [8M] | Wqb,Wkb,Wvb,Wob [1M each] | Qb,Kb,Vb [8M each] | AW [8M]  -> ~92MB
  const size_t nX = (size_t)NB * SEQ * DMODEL;       // 8,388,608
  const size_t nW = (size_t)DMODEL * DMODEL;         // 1,048,576
  __bf16* Xb  = (__bf16*)d_ws;
  __bf16* Wqb = Xb + nX;
  __bf16* Wkb = Wqb + nW;
  __bf16* Wvb = Wkb + nW;
  __bf16* Wob = Wvb + nW;
  __bf16* Qb  = Wob + nW;
  __bf16* Kb  = Qb + nX;
  __bf16* Vb  = Kb + nX;
  __bf16* AW  = Vb + nX;

  // one-time fp32 -> bf16 conversion (~100MB traffic, ~us at 23.3 TB/s)
  cvt_bf16_kernel<<<dim3((unsigned)(nX / 8 / 256)), 256, 0, stream>>>(x, Xb, (int)nX);
  cvt_bf16_kernel<<<dim3((unsigned)(nW / 8 / 256)), 256, 0, stream>>>(Wq, Wqb, (int)nW);
  cvt_bf16_kernel<<<dim3((unsigned)(nW / 8 / 256)), 256, 0, stream>>>(Wk, Wkb, (int)nW);
  cvt_bf16_kernel<<<dim3((unsigned)(nW / 8 / 256)), 256, 0, stream>>>(Wv, Wvb, (int)nW);
  cvt_bf16_kernel<<<dim3((unsigned)(nW / 8 / 256)), 256, 0, stream>>>(Wo, Wob, (int)nW);

  dim3 gemm_grid(NB * SEQ / BM, DMODEL / BN);        // (64, 8)
  gemm_bf16_kernel<true><<<gemm_grid, 256, 0, stream>>>(Xb, Wqb, Qb, nullptr, 1);
  gemm_bf16_kernel<true><<<gemm_grid, 256, 0, stream>>>(Xb, Wkb, Kb, nullptr, 1);
  gemm_bf16_kernel<true><<<gemm_grid, 256, 0, stream>>>(Xb, Wvb, Vb, nullptr, 0);

  flash_attn_kernel<<<dim3(SEQ / QT, NH, NB), 128, 0, stream>>>(Qb, Kb, Vb, AW);

  gemm_bf16_kernel<false><<<gemm_grid, 256, 0, stream>>>(AW, Wob, nullptr, out, 0);
}
